// DeformConv2d_7954279432832
// MI455X (gfx1250) — compile-verified, matrix operand order checked by reading, not run on hardware
//
#include <hip/hip_runtime.h>

typedef __attribute__((ext_vector_type(16))) _Float16 v16h;
typedef __attribute__((ext_vector_type(8)))  float    v8f;
typedef __attribute__((ext_vector_type(4)))  unsigned int u32x4;
typedef __attribute__((ext_vector_type(8)))  int      i32x8;
typedef __attribute__((ext_vector_type(4)))  int      i32x4;

#define B_    8
#define C_    64
#define H_    96
#define W_    96
#define OUT_  64
#define KTOT  5184          // 576 * 9
#define KT    162           // KTOT / 32
#define HW    9216          // 96*96
#define HO    94
#define XOELEMS (576*96*96) // 5,308,416

// workspace layout (bytes)
#define WPACK_DW  (KT*4*32*8)            // 165,888 dwords = 663,552 B
#define OFF_WPACK 0
#define OFF_COF   (WPACK_DW*4)           // 663,552
#define OFF_XO    (OFF_COF + KTOT*4)     // 684,288 (256B aligned)
// xo region: XOELEMS f16 + slack for guarded-but-unstored edge reads

// ---------------------------------------------------------------------------
// im2col K-offset table: k = cc*9 + (ky*3+kx)  ->  cc*HW + ky*96 + kx
__global__ void cof_kernel(int* __restrict__ cof) {
    int k = blockIdx.x * blockDim.x + threadIdx.x;
    if (k < KTOT) {
        int cc = k / 9, j = k - cc * 9;
        cof[k] = cc * HW + (j / 3) * W_ + (j % 3);
    }
}

// ---------------------------------------------------------------------------
// Pack conv_w (64 x 5184 f32) into f16 WMMA A-fragment order:
// [ktile][mtile][lane][vgpr] dwords; lane -> M=lane%16, g=lane/16,
// vgpr v -> K = 2v+8g (v<4) or 16+2(v-4)+8g (v>=4), pair (K,K+1) packed lo|hi.
__global__ void wpack_kernel(const float* __restrict__ w, unsigned int* __restrict__ wp) {
    int idx  = blockIdx.x * 256 + threadIdx.x;      // < WPACK_DW
    int v    = idx & 7;
    int lane = (idx >> 3) & 31;
    int mt   = (idx >> 8) & 3;
    int kt   = idx >> 10;
    int m    = mt * 16 + (lane & 15);
    int g    = lane >> 4;
    int kk   = (v < 4) ? (2 * v + 8 * g) : (16 + 2 * (v - 4) + 8 * g);
    int k    = kt * 32 + kk;                         // always even, k+1 <= 5183
    _Float16 e0 = (_Float16)w[m * KTOT + k];
    _Float16 e1 = (_Float16)w[m * KTOT + k + 1];
    unsigned int p = (unsigned int)__builtin_bit_cast(unsigned short, e0) |
                     ((unsigned int)__builtin_bit_cast(unsigned short, e1) << 16);
    wp[idx] = p;
}

// ---------------------------------------------------------------------------
// Deformable bilinear gather (exact f32 math, f16 store), one batch.
// Replicates reference's scrambled permute/reshape index math analytically.
__global__ void gather_kernel(const float* __restrict__ x, const float* __restrict__ off,
                              _Float16* __restrict__ xo, int b) {
    int i = blockIdx.x * 256 + threadIdx.x;          // < XOELEMS
    int CC  = i / HW;
    int rem = i - CC * HW;
    int Y   = rem / W_;
    int X   = rem - Y * W_;
    // val-reshape scramble: (CC,Y,X) -> (c, y, x, n)
    int c    = CC / 9;
    int m    = CC - c * 9;
    int flat = m * HW + Y * W_ + X;
    int y    = flat / 864;
    int r2   = flat - y * 864;
    int xb   = r2 / 9;
    int n    = r2 - xb * 9;
    // p-reshape scramble (shared parts)
    int comp = xb & 1;
    int xx   = (xb >> 1) + 48 * (y & 1);
    // qx from p channel n
    int s1   = n >> 1;
    int g1   = comp ? (s1 % 3) : (s1 / 3);
    int yy1  = (y >> 1) + 48 * (n & 1);
    int ch1  = (n & ~1) + comp;
    float qx = (float)g1 + off[(((size_t)b * 18 + ch1) * H_ + yy1) * W_ + xx];
    // qy from p channel 9+n
    int s2   = (9 + n) >> 1;
    int g2   = comp ? (s2 % 3) : (s2 / 3);
    int yy2  = (y >> 1) + 48 * ((n + 1) & 1);
    int ch2  = ((9 + n) & ~1) + comp;
    float qy = (float)g2 + off[(((size_t)b * 18 + ch2) * H_ + yy2) * W_ + xx];
    // clip to padded image [0, 97]
    qx = fminf(fmaxf(qx, 0.f), 97.f);
    qy = fminf(fmaxf(qy, 0.f), 97.f);
    int   x0 = (int)floorf(qx); int x1 = min(x0 + 1, 97);
    int   y0 = (int)floorf(qy); int y1 = min(y0 + 1, 97);
    float wx = qx - (float)x0;
    float wy = qy - (float)y0;
    const float* xc = x + ((size_t)b * C_ + c) * HW;
    auto samp = [&](int rr, int cl) -> float {       // padded image, PAD=1
        rr -= 1; cl -= 1;
        return ((unsigned)rr < 96u && (unsigned)cl < 96u) ? xc[rr * W_ + cl] : 0.f;
    };
    float v00 = samp(y0, x0), v01 = samp(y0, x1);
    float v10 = samp(y1, x0), v11 = samp(y1, x1);
    float val = (v00 * (1.f - wx) + v01 * wx) * (1.f - wy) +
                (v10 * (1.f - wx) + v11 * wx) * wy;
    xo[i] = (_Float16)val;
}

// ---------------------------------------------------------------------------
// Implicit-im2col GEMM via v_wmma_f32_16x16x32_f16.
// cof table is DMA'd into LDS by the Tensor Data Mover (one TDM op / block).
// Block = 8 waves: 4 M-tiles (all 64 outputs) x 2 N-tiles (16 pixels each).
// Grid: (3 ox-tile-pairs, 94 oy rows) per batch.
__global__ void __launch_bounds__(256) conv_kernel(const unsigned int* __restrict__ wp,
                                                   const int* __restrict__ cof,
                                                   const unsigned short* __restrict__ xo,
                                                   const float* __restrict__ bias,
                                                   float* __restrict__ out, int b) {
    __shared__ __align__(16) int cofLds[KTOT];

    // --- TDM async tensor load: cof[5184] (1D, 4B elems) -> LDS -------------
    if (threadIdx.x < 32) {                      // one wave issues the DMA
        unsigned long long ga = (unsigned long long)(const void*)cof;
        unsigned lds_a = (unsigned)(unsigned long long)(void*)&cofLds[0];
        u32x4 g0;
        g0.x = 1u;                                // D# group0: count=1 (user)
        g0.y = lds_a;                             // lds_addr
        g0.z = (unsigned)ga;                      // global_addr[31:0]
        g0.w = (unsigned)((ga >> 32) & 0x1FFFFFFu) | (2u << 30); // [56:32] | type=2
        i32x8 g1;
        g1[0] = 0x00020000;                       // data_size=4B; no multicast/pad
        g1[1] = (int)((KTOT & 0xFFFF) << 16);     // tensor_dim0[15:0]=5184
        g1[2] = (int)(1 << 16);                   // tensor_dim0[31:16]=0 | tensor_dim1=1
        g1[3] = (int)((KTOT & 0xFFFF) << 16);     // tile_dim0=5184
        g1[4] = 1;                                // tile_dim1=1, tile_dim2=0
        g1[5] = KTOT;                             // tensor_dim0_stride[31:0]
        g1[6] = (int)((KTOT & 0xFFFF) << 16);     // stride0 hi=0 | dim1_stride lo
        g1[7] = 0;
        i32x4 gz = {0, 0, 0, 0};                  // groups 2/3: dims unused
#if defined(__clang_major__) && (__clang_major__ >= 23)
        i32x8 gz8 = {0, 0, 0, 0, 0, 0, 0, 0};
        __builtin_amdgcn_tensor_load_to_lds(g0, g1, gz, gz, gz8, 0);
#else
        __builtin_amdgcn_tensor_load_to_lds(g0, g1, gz, gz, 0);
#endif
        __builtin_amdgcn_s_wait_tensorcnt(0);
    }
    __syncthreads();

    const int lane = threadIdx.x & 31;
    const int wv   = threadIdx.x >> 5;
    const int mt   = wv & 3;
    const int nt   = wv >> 2;
    const int oy   = blockIdx.y;
    const int ox0  = (blockIdx.x * 2 + nt) * 16;
    const int col  = lane & 15;
    const int g    = lane >> 4;
    const int pixbase = oy * W_ + ox0 + col;     // + cof[k] = xo element index

    v8f acc = {0.f, 0.f, 0.f, 0.f, 0.f, 0.f, 0.f, 0.f};

    #pragma unroll 2
    for (int kt = 0; kt < KT; ++kt) {
        // A fragment: pre-swizzled, 32B contiguous per lane
        union { uint4 q[2]; v16h h; } a;
        const uint4* ap = (const uint4*)(wp + ((size_t)(kt * 4 + mt) * 32 + lane) * 8);
        a.q[0] = ap[0];
        a.q[1] = ap[1];
        // B fragment: lane = pixel column, per-VGPR packed K pair via cof table
        union { unsigned int u[8]; v16h h; } bm;
        #pragma unroll
        for (int v = 0; v < 8; ++v) {
            int kk = (v < 4) ? (2 * v + 8 * g) : (16 + 2 * (v - 4) + 8 * g);
            int k0 = kt * 32 + kk;
            int cf0 = cofLds[k0];
            int cf1 = cofLds[k0 + 1];
            unsigned int e0 = xo[cf0 + pixbase];
            unsigned int e1 = xo[cf1 + pixbase];
            bm.u[v] = e0 | (e1 << 16);
        }
        acc = __builtin_amdgcn_wmma_f32_16x16x32_f16(false, a.h, false, bm.h,
                                                     (short)0, acc, false, false);
    }

    // C/D layout: VGPR r holds M = r + 8*g, N = lane%16
    const int ox = ox0 + col;
    if (ox < HO) {
        const int obase = mt * 16 + 8 * g;
        #pragma unroll
        for (int r = 0; r < 8; ++r) {
            int o = obase + r;
            out[(((size_t)b * OUT_ + o) * HO + oy) * HO + ox] = acc[r] + bias[o];
        }
    }
}

// ---------------------------------------------------------------------------
extern "C" void kernel_launch(void* const* d_in, const int* in_sizes, int n_in,
                              void* d_out, int out_size, void* d_ws, size_t ws_size,
                              hipStream_t stream) {
    const float* x   = (const float*)d_in[0];
    const float* off = (const float*)d_in[1];
    const float* cw  = (const float*)d_in[2];
    const float* cb  = (const float*)d_in[3];
    float* out       = (float*)d_out;

    unsigned char* ws = (unsigned char*)d_ws;
    unsigned int* wp  = (unsigned int*)(ws + OFF_WPACK);
    int*          cof = (int*)(ws + OFF_COF);
    _Float16*     xo  = (_Float16*)(ws + OFF_XO);

    cof_kernel  <<<(KTOT + 255) / 256, 256, 0, stream>>>(cof);
    wpack_kernel<<<WPACK_DW / 256,     256, 0, stream>>>(cw, wp);

    for (int b = 0; b < B_; ++b) {
        // stream order serializes: conv(b) finishes before gather(b+1) reuses xo
        gather_kernel<<<XOELEMS / 256, 256, 0, stream>>>(x, off, xo, b);
        conv_kernel  <<<dim3(3, HO),   256, 0, stream>>>(wp, cof,
                                                         (const unsigned short*)xo,
                                                         cb, out, b);
    }
}
